// GraphCast_29729763623324
// MI455X (gfx1250) — matrix operand order repaired and checked by reading, not compiled, not run on hardware
//
#include <hip/hip_runtime.h>

// ---------------------------------------------------------------------------
// CDNA5 / gfx1250 GraphCast pipeline.
// All GEMMs: v_wmma_f32_16x16x32_bf16 (wave32 WMMA), fp32 accumulate.
// All activations stored bf16 (converted once in GEMM epilogues), so the
// GEMM A-path is a pure copy -> staged with CDNA5 async-to-LDS loads
// (global_load_async_to_lds_b128 / s_wait_asynccnt), 4-buffer rotation,
// distance-2 prefetch. B weights pre-transposed to bf16 col-major (L2).
// ---------------------------------------------------------------------------

#define USE_ASYNC_LDS 1

typedef __attribute__((ext_vector_type(16))) __bf16 v16bf;
typedef __attribute__((ext_vector_type(8)))  float  v8f;
typedef unsigned int u32x4 __attribute__((ext_vector_type(4)));
typedef unsigned int u32x8 __attribute__((ext_vector_type(8)));

union BFrag { u32x8 u; v16bf v; };                  // 16 bf16 = 32B

__device__ __forceinline__ unsigned short f2bf(float f) {
  unsigned u = __float_as_uint(f);
  u += 0x7FFFu + ((u >> 16) & 1u);                  // round-to-nearest-even
  return (unsigned short)(u >> 16);
}
__device__ __forceinline__ unsigned pk2(float a, float b) {
  return (unsigned)f2bf(a) | ((unsigned)f2bf(b) << 16);
}
__device__ __forceinline__ float b2f(unsigned short u) {
  return __uint_as_float((unsigned)u << 16);
}

#if USE_ASYNC_LDS
__device__ __forceinline__ void async_cp16(unsigned lds_off, const void* g) {
  // ASYNCcnt-tracked DMA: 16B global -> LDS, per-lane addressed
  asm volatile("global_load_async_to_lds_b128 %0, %1, off"
               :: "v"(lds_off), "v"(g) : "memory");
}
__device__ __forceinline__ void wait_async(int rem) {
  if (rem >= 2)      asm volatile("s_wait_asynccnt 0x2" ::: "memory");
  else if (rem == 1) asm volatile("s_wait_asynccnt 0x1" ::: "memory");
  else               asm volatile("s_wait_asynccnt 0x0" ::: "memory");
}
#endif

static constexpr int LDIM = 512;

// ---------------------------------------------------------------------------
// Main WMMA GEMM: out[N,512] = EPI( sum_c X_c[idx_c][:,Kp_c] @ W_c + bias )
//   block = 64 rows x 512 cols, 8 waves; each wave = 64x64 (4x4 WMMA tiles).
//   Invariants (host-side packing): every chunk has K == Kp == stride
//   (multiple of 64); X rows / idx valid to rup64(N). Inner loop branch-free.
// EPI: 0 = SiLU store (hidden layer), 1 = LayerNorm*g+be (+resid) store.
// All activations bf16; accumulate fp32.
// ---------------------------------------------------------------------------
struct GemmP {
  const unsigned short* X[3];
  const int*            idx[3];
  int Kp[3];                  // K == stride, multiple of 64
  int nch;
  const unsigned short* Wt;   // [512][Ktot] bf16, col-major (transposed weight)
  int Ktot;
  const float* bias;
  const float* gamma;
  const float* beta;
  const unsigned short* resid;
  unsigned short* out;
  int N;
};

template <int EPI>
__global__ __launch_bounds__(256)
void gemm_mlp(GemmP p)
{
#if USE_ASYNC_LDS
  __shared__ __align__(16) unsigned short ldsA[4][64 * 32];  // 4 x (64 rows x 32 K)
#else
  __shared__ __align__(16) unsigned short ldsA[2][64 * 32];
#endif
  __shared__ float lsum[64][8], lsq[64][8], smu[64], srs[64];

  const int t  = threadIdx.x;
  const int w  = t >> 5;          // wave 0..7 -> col group of 64
  const int l  = t & 31;
  const int li = l & 15;
  const int hi = l >> 4;
  const long row0 = (long)blockIdx.x * 64;

  v8f acc[4][4];
  #pragma unroll
  for (int m = 0; m < 4; ++m)
    #pragma unroll
    for (int n = 0; n < 4; ++n)
      #pragma unroll
      for (int e = 0; e < 8; ++e) acc[m][n][e] = 0.f;

  // A-stage assignment: thread -> (row t>>2, 8 consecutive bf16 k)
  const int  ar   = t >> 2;
  const int  ak   = (t & 3) << 3;
  const long arow = row0 + ar;

  u32x4 bl[2][4], bh[2][4];
  auto loadB = [&](int set, long kpos) {
    #pragma unroll
    for (int nt = 0; nt < 4; ++nt) {
      const unsigned short* wp =
          p.Wt + (long)(w * 64 + nt * 16 + li) * p.Ktot + kpos + hi * 8;
      bl[set][nt] = *(const u32x4*)wp;
      bh[set][nt] = *(const u32x4*)(wp + 16);
    }
  };
  auto mmaAB = [&](int set, const unsigned short* lbase) {
    // A fragments (ISA 7.12.2): lane li = row; lanes 0-15: K {0..7,16..23},
    // lanes 16-31: K {8..15,24..31}
    BFrag a[4];
    #pragma unroll
    for (int mt = 0; mt < 4; ++mt) {
      const u32x4* pa = (const u32x4*)(lbase + (mt * 16 + li) * 32 + hi * 8);
      a[mt].u = __builtin_shufflevector(pa[0], pa[2], 0, 1, 2, 3, 4, 5, 6, 7);
    }
    BFrag bf[4];
    #pragma unroll
    for (int nt = 0; nt < 4; ++nt)
      bf[nt].u = __builtin_shufflevector(bl[set][nt], bh[set][nt],
                                         0, 1, 2, 3, 4, 5, 6, 7);
    #pragma unroll
    for (int nt = 0; nt < 4; ++nt)
      #pragma unroll
      for (int mt = 0; mt < 4; ++mt)
        acc[mt][nt] = __builtin_amdgcn_wmma_f32_16x16x32_bf16(
            false, a[mt].v, false, bf[nt].v, (short)0, acc[mt][nt], false, false);
  };

  int koff = 0;
  for (int c = 0; c < p.nch; ++c) {
    const int Kp = p.Kp[c];
    const long grow = p.idx[c] ? (long)p.idx[c][arow] : arow;   // fused gather
    const unsigned short* xr = p.X[c] + grow * (long)Kp + ak;

#if USE_ASYNC_LDS
    if (c) __syncthreads();       // drain readers before reusing buffers
    auto issueStage = [&](int buf, int kk) {
      async_cp16((unsigned)(uintptr_t)&ldsA[buf][ar * 32 + ak], xr + kk);
    };
    const int nhs = Kp >> 5;      // half-steps (>= 2, even)
    issueStage(0, 0);
    issueStage(1, 32);
    loadB(0, koff);
    for (int hs = 0; hs < nhs; ++hs) {
      if (hs + 2 < nhs) issueStage((hs + 2) & 3, (hs + 2) << 5);
      wait_async(nhs - 1 - hs);                  // current stage complete
      __syncthreads();
      if (hs + 1 < nhs) loadB((hs + 1) & 1, (long)koff + ((hs + 1) << 5));
      mmaAB(hs & 1, &ldsA[hs & 3][0]);
    }
#else
    u32x4 va[2];
    auto stageLoad = [&](int set, int kk) { va[set] = *(const u32x4*)(xr + kk); };
    auto commitA   = [&](int set, int buf) {
      *(u32x4*)&ldsA[buf][ar * 32 + ak] = va[set];
    };
    stageLoad(0, 0);
    loadB(0, koff);
    for (int k0 = 0; k0 < Kp; k0 += 64) {
      commitA(0, 0);
      __syncthreads();
      stageLoad(1, k0 + 32);
      loadB(1, (long)koff + k0 + 32);
      mmaAB(0, &ldsA[0][0]);
      commitA(1, 1);
      __syncthreads();
      if (k0 + 64 < Kp) { stageLoad(0, k0 + 64); loadB(0, (long)koff + k0 + 64); }
      mmaAB(1, &ldsA[1][0]);
    }
#endif
    koff += Kp;
  }

  // ---- bias (C layout: VGPR j -> row j+8*hi; col = lane&15) ----
  #pragma unroll
  for (int mt = 0; mt < 4; ++mt)
    #pragma unroll
    for (int nt = 0; nt < 4; ++nt) {
      const float b = p.bias[w * 64 + nt * 16 + li];
      #pragma unroll
      for (int j = 0; j < 8; ++j) acc[mt][nt][j] += b;
    }

  if constexpr (EPI == 0) {
    // SiLU, bf16 store
    for (int mt = 0; mt < 4; ++mt)
      for (int j = 0; j < 8; ++j) {
        const long row = row0 + mt * 16 + hi * 8 + j;
        if (row < p.N) {
          unsigned short* op = p.out + row * (long)LDIM + w * 64 + li;
          #pragma unroll
          for (int nt = 0; nt < 4; ++nt) {
            float x = acc[mt][nt][j];
            op[nt * 16] = f2bf(x * (1.f / (1.f + __expf(-x))));
          }
        }
      }
  } else {
    // LayerNorm over the 512-wide row: shfl-reduce + tiny LDS cross-wave.
    for (int mt = 0; mt < 4; ++mt)
      for (int j = 0; j < 8; ++j) {
        float s = 0.f, q = 0.f;
        #pragma unroll
        for (int nt = 0; nt < 4; ++nt) { float x = acc[mt][nt][j]; s += x; q += x * x; }
        for (int o = 1; o < 16; o <<= 1) { s += __shfl_xor(s, o, 32); q += __shfl_xor(q, o, 32); }
        if (li == 0) { const int rt = mt * 16 + hi * 8 + j; lsum[rt][w] = s; lsq[rt][w] = q; }
      }
    __syncthreads();
    if (t < 64) {
      float s = 0.f, q = 0.f;
      #pragma unroll
      for (int ww = 0; ww < 8; ++ww) { s += lsum[t][ww]; q += lsq[t][ww]; }
      const float mu  = s * (1.f / 512.f);
      const float var = q * (1.f / 512.f) - mu * mu;
      smu[t] = mu; srs[t] = rsqrtf(var + 1e-5f);
    }
    __syncthreads();
    for (int mt = 0; mt < 4; ++mt)
      for (int j = 0; j < 8; ++j) {
        const int  rt  = mt * 16 + hi * 8 + j;
        const long row = row0 + rt;
        if (row < p.N) {
          const float mu = smu[rt], rs = srs[rt];
          #pragma unroll
          for (int nt = 0; nt < 4; ++nt) {
            const int cc = w * 64 + nt * 16 + li;
            float x = (acc[mt][nt][j] - mu) * rs * p.gamma[cc] + p.beta[cc];
            if (p.resid) x += b2f(p.resid[row * (long)LDIM + cc]);
            p.out[row * (long)LDIM + cc] = f2bf(x);
          }
        }
      }
  }
}

// ---------------------------------------------------------------------------
// out_mlp layer 2: bf16 [NG,512] @ bf16 [512,64] + b + residual(x fp32)
// -> d_out fp32 [64][NG]. H is bf16 so A fragments are direct b128 loads.
// ---------------------------------------------------------------------------
__global__ __launch_bounds__(256)
void gemm_out64(const unsigned short* __restrict__ H,
                const unsigned short* __restrict__ Wt,
                const float* __restrict__ b, const float* __restrict__ xin,
                float* __restrict__ out, int N)
{
  const int t = threadIdx.x, w = t >> 5, l = t & 31, li = l & 15, hi = l >> 4;
  const long row0 = (long)blockIdx.x * 128;
  const long arow = row0 + w * 16 + li;
  const unsigned short* hrow = H + arow * (long)512 + hi * 8;

  v8f acc[4];
  #pragma unroll
  for (int n = 0; n < 4; ++n)
    #pragma unroll
    for (int e = 0; e < 8; ++e) acc[n][e] = 0.f;

  u32x4 al[2], ah[2], bl[2][4], bh[2][4];
  auto loadH = [&](int set, int k0) {
    al[set] = *(const u32x4*)(hrow + k0);
    ah[set] = *(const u32x4*)(hrow + k0 + 16);
  };
  auto loadB = [&](int set, int k0) {
    #pragma unroll
    for (int nt = 0; nt < 4; ++nt) {
      const unsigned short* wp = Wt + (long)(nt * 16 + li) * 512 + k0 + hi * 8;
      bl[set][nt] = *(const u32x4*)wp;
      bh[set][nt] = *(const u32x4*)(wp + 16);
    }
  };
  auto mma = [&](int set) {
    BFrag a; a.u = __builtin_shufflevector(al[set], ah[set], 0, 1, 2, 3, 4, 5, 6, 7);
    #pragma unroll
    for (int nt = 0; nt < 4; ++nt) {
      BFrag bf; bf.u = __builtin_shufflevector(bl[set][nt], bh[set][nt],
                                               0, 1, 2, 3, 4, 5, 6, 7);
      acc[nt] = __builtin_amdgcn_wmma_f32_16x16x32_bf16(
          false, a.v, false, bf.v, (short)0, acc[nt], false, false);
    }
  };

  loadH(0, 0); loadB(0, 0);
  for (int k0 = 0; k0 < 512; k0 += 64) {
    loadH(1, k0 + 32); loadB(1, k0 + 32);
    mma(0);
    if (k0 + 64 < 512) { loadH(0, k0 + 64); loadB(0, k0 + 64); }
    mma(1);
  }
  #pragma unroll
  for (int nt = 0; nt < 4; ++nt) {
    const int cc = nt * 16 + li;
    for (int j = 0; j < 8; ++j) {
      const long r = row0 + w * 16 + hi * 8 + j;
      if (r < N) out[(long)cc * N + r] = acc[nt][j] + b[cc] + xin[(long)cc * N + r];
    }
  }
}

// ---------------------------------------------------------------------------
// Utility kernels
// ---------------------------------------------------------------------------
__global__ void k_conv_w(const float* __restrict__ W, int rowOff, int rows, int ncol,
                         unsigned short* __restrict__ Wt, int ktot, int koff, int kpad)
{
  long i   = (long)blockIdx.x * 256 + threadIdx.x;
  long tot = (long)ncol * kpad;
  if (i >= tot) return;
  int n = (int)(i / kpad), k = (int)(i % kpad);
  float v = (k < rows) ? W[(long)(rowOff + k) * ncol + n] : 0.f;
  Wt[(long)n * ktot + koff + k] = f2bf(v);
}

__global__ void k_zero4(float4* p, long n4)
{ long i = (long)blockIdx.x * 256 + threadIdx.x;
  if (i < n4) p[i] = make_float4(0.f, 0.f, 0.f, 0.f); }

// packed-bf16 in-place add: a += b (2 elems per uint)
__global__ void k_addbf(unsigned* a, const unsigned* b, long n2)
{
  long i = (long)blockIdx.x * 256 + threadIdx.x;
  if (i < n2) {
    unsigned x = a[i], y = b[i];
    float lo = b2f((unsigned short)x) + b2f((unsigned short)y);
    float hf = b2f((unsigned short)(x >> 16)) + b2f((unsigned short)(y >> 16));
    a[i] = pk2(lo, hf);
  }
}

// fp32 buffer -> bf16 buffer
__global__ void k_f2b(const float* __restrict__ src, unsigned short* __restrict__ dst,
                      long n)
{ long i = (long)blockIdx.x * 256 + threadIdx.x;
  if (i < n) dst[i] = f2bf(src[i]); }

// copy fp32 src[N,Ks] into bf16 dst[:, coff:coff+Ks] (stride elems per row)
__global__ void k_copy_cols(const float* __restrict__ src,
                            unsigned short* __restrict__ dst,
                            long N, int Ks, int stride, int coff)
{
  long i = (long)blockIdx.x * 256 + threadIdx.x;
  if (i < N * Ks) {
    long r = i / Ks; int c = (int)(i % Ks);
    dst[r * stride + coff + c] = f2bf(src[i]);
  }
}

__global__ void k_pad_idx(const int* __restrict__ src, int* __restrict__ dst,
                          long n, long np)
{ long i = (long)blockIdx.x * 256 + threadIdx.x;
  if (i < np) dst[i] = (i < n) ? src[i] : 0; }

// scatter-add bf16 msg into fp32 agg
__global__ void k_scatter(const unsigned short* __restrict__ msg,
                          const int* __restrict__ recv,
                          float* __restrict__ agg, long n)
{
  long i = (long)blockIdx.x * 256 + threadIdx.x;
  if (i < n) {
    long e = i >> 9; int c = (int)(i & 511);
    atomicAdd(agg + (long)recv[e] * 512 + c, b2f(msg[i]));
  }
}

// x [64][NG] fp32 -> bf16 dst[n][c] at stride (transpose), n < NG
__global__ void k_xt(const float* __restrict__ x, unsigned short* __restrict__ xt,
                     int ng, int stride)
{
  long i = (long)blockIdx.x * 256 + threadIdx.x;
  if (i < (long)ng * 64) {
    int n = (int)(i >> 6), c = (int)(i & 63);
    xt[(long)n * stride + c] = f2bf(x[(long)c * ng + n]);
  }
}

// ---------------------------------------------------------------------------
// Host orchestration
// ---------------------------------------------------------------------------
struct WtMat { const unsigned short* w; int ktot; };
struct Chunk { const unsigned short* X; const int* idx; int Kp; };
struct P6    { const float *b1, *b2, *be, *g, *w1, *w2; };

static inline long rup64(long k) { return (k + 63) & ~(long)63; }
static inline unsigned gsz(long n) { return (unsigned)((n + 255) / 256); }

static void launch_gemm(hipStream_t s, int epi, const Chunk* ch, int nch,
                        WtMat W, const float* bias, const float* g, const float* be,
                        const unsigned short* resid, unsigned short* out, int N)
{
  GemmP p{};
  for (int i = 0; i < nch; ++i) {
    p.X[i] = ch[i].X; p.idx[i] = ch[i].idx; p.Kp[i] = ch[i].Kp;
  }
  p.nch = nch; p.Wt = W.w; p.Ktot = W.ktot;
  p.bias = bias; p.gamma = g; p.beta = be; p.resid = resid; p.out = out; p.N = N;
  dim3 grid((unsigned)((N + 63) / 64));
  if (epi == 0) gemm_mlp<0><<<grid, 256, 0, s>>>(p);
  else          gemm_mlp<1><<<grid, 256, 0, s>>>(p);
}

extern "C" void kernel_launch(void* const* d_in, const int* in_sizes, int n_in,
                              void* d_out, int out_size, void* d_ws, size_t ws_size,
                              hipStream_t stream)
{
  (void)in_sizes; (void)n_in; (void)out_size; (void)ws_size;
  const int  NG = 29040, NM = 10242, EG = 80000, EM = 65536, ED = 87120;
  const long NGp = rup64(NG), NMp = rup64(NM);
  const long EGp = rup64(EG), EMp = rup64(EM), EDp = rup64(ED);

  // -------- inputs (setup_inputs order) --------
  const float* x      = (const float*)d_in[0];   // [64][NG]
  const float* grid_s = (const float*)d_in[1];
  const float* mesh_s = (const float*)d_in[2];
  const float* eg_s   = (const float*)d_in[3];
  const float* em_s   = (const float*)d_in[4];
  const float* ed_s   = (const float*)d_in[5];
  const int* g2m_s = (const int*)d_in[6],  *g2m_r = (const int*)d_in[7];
  const int* m2m_s = (const int*)d_in[8],  *m2m_r = (const int*)d_in[9];
  const int* m2g_s = (const int*)d_in[10], *m2g_r = (const int*)d_in[11];

  // -------- params: JAX pytree order (sorted dict keys; inner b1,b2,be,g,w1,w2)
  auto get6 = [&](int i) -> P6 {
    return P6{ (const float*)d_in[i], (const float*)d_in[i+1], (const float*)d_in[i+2],
               (const float*)d_in[i+3], (const float*)d_in[i+4], (const float*)d_in[i+5] };
  };
  P6 eg2m_enc = get6(12), em2g_enc = get6(18), em2m_enc = get6(24);
  P6 g2m_edge = get6(30), g2m_node_grid = get6(36), g2m_node_mesh = get6(42);
  P6 grid_enc = get6(48), m2g_edge = get6(54), m2g_node = get6(60), mesh_enc = get6(66);
  const float* out_b1 = (const float*)d_in[72];
  const float* out_b2 = (const float*)d_in[73];
  const float* out_w1 = (const float*)d_in[74];
  const float* out_w2 = (const float*)d_in[75];
  P6 proc_edge[4], proc_node[4];
  for (int i = 0; i < 4; ++i) { proc_edge[i] = get6(76 + 6*i); proc_node[i] = get6(100 + 6*i); }

  // -------- workspace bump allocator --------
  size_t off = 0;
  auto alloc = [&](size_t bytes) -> void* {
    void* p = (char*)d_ws + off;
    off += (bytes + 255) & ~(size_t)255;
    return p;
  };
  typedef unsigned short u16;
  // bf16 packed inputs (row- and K-padded)
  u16* xgp = (u16*)alloc((size_t)NGp * 128 * 2);   // [x | grid_s | 0] K=128
  u16* msp = (u16*)alloc((size_t)NMp * 64 * 2);
  u16* egp = (u16*)alloc((size_t)EGp * 64 * 2);
  u16* emp = (u16*)alloc((size_t)EMp * 64 * 2);
  u16* edp = (u16*)alloc((size_t)EDp * 64 * 2);
  int* m2g_s_p = (int*)alloc((size_t)EDp * 4);
  int* m2g_r_p = (int*)alloc((size_t)EDp * 4);
  // bf16 operand buffers (row-padded)
  u16* vG   = (u16*)alloc((size_t)NGp * 512 * 2);
  u16* vM   = (u16*)alloc((size_t)NMp * 512 * 2);
  u16* eMb  = (u16*)alloc((size_t)EMp * 512 * 2);
  u16* eXb  = (u16*)alloc((size_t)EDp * 512 * 2);  // eG, later reused for eD
  u16* msg  = (u16*)alloc((size_t)EDp * 512 * 2);
  u16* hb   = (u16*)alloc((size_t)EDp * 512 * 2);  // hidden-layer scratch
  u16* aggb = (u16*)alloc((size_t)NGp * 512 * 2);
  float* agg = (float*)alloc((size_t)NGp * 512 * 4);   // fp32 scatter target

  // -------- pack padded inputs (zero-fill then copy) --------
  k_zero4<<<gsz(NGp * 16), 256, 0, stream>>>((float4*)xgp, NGp * 16);
  k_zero4<<<gsz(NMp * 8),  256, 0, stream>>>((float4*)msp, NMp * 8);
  k_zero4<<<gsz(EGp * 8),  256, 0, stream>>>((float4*)egp, EGp * 8);
  k_zero4<<<gsz(EMp * 8),  256, 0, stream>>>((float4*)emp, EMp * 8);
  k_zero4<<<gsz(EDp * 8),  256, 0, stream>>>((float4*)edp, EDp * 8);
  k_xt<<<gsz((long)NG * 64), 256, 0, stream>>>(x, xgp, NG, 128);
  k_copy_cols<<<gsz((long)NG * 3), 256, 0, stream>>>(grid_s, xgp, NG, 3, 128, 64);
  k_copy_cols<<<gsz((long)NM * 3), 256, 0, stream>>>(mesh_s, msp, NM, 3, 64, 0);
  k_copy_cols<<<gsz((long)EG * 4), 256, 0, stream>>>(eg_s, egp, EG, 4, 64, 0);
  k_copy_cols<<<gsz((long)EM * 4), 256, 0, stream>>>(em_s, emp, EM, 4, 64, 0);
  k_copy_cols<<<gsz((long)ED * 4), 256, 0, stream>>>(ed_s, edp, ED, 4, 64, 0);
  k_pad_idx<<<gsz(EDp), 256, 0, stream>>>(m2g_s, m2g_s_p, ED, EDp);
  k_pad_idx<<<gsz(EDp), 256, 0, stream>>>(m2g_r, m2g_r_p, ED, EDp);

  // -------- weight convert: fp32 [K][ncol] -> bf16 col-major [ncol][Kpad] ----
  auto convw = [&](const float* W, int ncol, const int* rows, int nch) -> WtMat {
    int ktot = 0;
    for (int i = 0; i < nch; ++i) ktot += (int)rup64(rows[i]);
    u16* Wt = (u16*)alloc((size_t)ncol * ktot * 2);
    int ko = 0, ro = 0;
    for (int i = 0; i < nch; ++i) {
      int kp = (int)rup64(rows[i]);
      long tot = (long)ncol * kp;
      k_conv_w<<<gsz(tot), 256, 0, stream>>>(W, ro, rows[i], ncol, Wt, ktot, ko, kp);
      ko += kp; ro += rows[i];
    }
    return WtMat{ Wt, ktot };
  };

  const int r1[1] = {512}, r2[2] = {512, 512}, r3[3] = {512, 512, 512};
  const int rgrd[2] = {64, 3}, rs3[1] = {3}, rs4[1] = {4};

  WtMat Wgrid1 = convw(grid_enc.w1, 512, rgrd, 2), Wgrid2 = convw(grid_enc.w2, 512, r1, 1);
  WtMat Wmesh1 = convw(mesh_enc.w1, 512, rs3, 1), Wmesh2 = convw(mesh_enc.w2, 512, r1, 1);
  WtMat Weg1 = convw(eg2m_enc.w1, 512, rs4, 1), Weg2 = convw(eg2m_enc.w2, 512, r1, 1);
  WtMat Wem1 = convw(em2m_enc.w1, 512, rs4, 1), Wem2 = convw(em2m_enc.w2, 512, r1, 1);
  WtMat Wed1 = convw(em2g_enc.w1, 512, rs4, 1), Wed2 = convw(em2g_enc.w2, 512, r1, 1);
  WtMat Wge1 = convw(g2m_edge.w1, 512, r3, 3), Wge2 = convw(g2m_edge.w2, 512, r1, 1);
  WtMat Wgm1 = convw(g2m_node_mesh.w1, 512, r2, 2), Wgm2 = convw(g2m_node_mesh.w2, 512, r1, 1);
  WtMat Wgg1 = convw(g2m_node_grid.w1, 512, r1, 1), Wgg2 = convw(g2m_node_grid.w2, 512, r1, 1);
  WtMat Wde1 = convw(m2g_edge.w1, 512, r3, 3), Wde2 = convw(m2g_edge.w2, 512, r1, 1);
  WtMat Wdn1 = convw(m2g_node.w1, 512, r2, 2), Wdn2 = convw(m2g_node.w2, 512, r1, 1);
  WtMat Wpe1[4], Wpe2[4], Wpn1[4], Wpn2[4];
  for (int i = 0; i < 4; ++i) {
    Wpe1[i] = convw(proc_edge[i].w1, 512, r3, 3); Wpe2[i] = convw(proc_edge[i].w2, 512, r1, 1);
    Wpn1[i] = convw(proc_node[i].w1, 512, r2, 2); Wpn2[i] = convw(proc_node[i].w2, 512, r1, 1);
  }
  WtMat Wo1 = convw(out_w1, 512, r1, 1), Wo2 = convw(out_w2, 64, r1, 1);

  // MLP = SiLU GEMM (layer1 -> hb) + LayerNorm GEMM (layer2 -> out, opt resid)
  auto mlp = [&](const Chunk* ch, int nch, WtMat W1, const float* b1,
                 WtMat W2, const float* b2, const float* g, const float* be,
                 const u16* resid, u16* out, int N) {
    launch_gemm(stream, 0, ch, nch, W1, b1, nullptr, nullptr, nullptr, hb, N);
    Chunk c2{hb, nullptr, 512};
    launch_gemm(stream, 1, &c2, 1, W2, b2, g, be, resid, out, N);
  };

  // ================= pipeline =================
  // --- encoders (packed single-chunk inputs) ---
  { Chunk c{xgp, nullptr, 128};
    mlp(&c, 1, Wgrid1, grid_enc.b1, Wgrid2, grid_enc.b2, grid_enc.g, grid_enc.be, nullptr, vG, NG); }
  { Chunk c{msp, nullptr, 64};
    mlp(&c, 1, Wmesh1, mesh_enc.b1, Wmesh2, mesh_enc.b2, mesh_enc.g, mesh_enc.be, nullptr, vM, NM); }
  { Chunk c{egp, nullptr, 64};
    mlp(&c, 1, Weg1, eg2m_enc.b1, Weg2, eg2m_enc.b2, eg2m_enc.g, eg2m_enc.be, nullptr, eXb, EG); }
  { Chunk c{emp, nullptr, 64};
    mlp(&c, 1, Wem1, em2m_enc.b1, Wem2, em2m_enc.b2, em2m_enc.g, em2m_enc.be, nullptr, eMb, EM); }

  // --- G2M: edge msg (gather fused into GEMM), segment-sum, node updates ---
  { Chunk c[3] = {{vG, g2m_s, 512}, {vM, g2m_r, 512}, {eXb, nullptr, 512}};
    mlp(c, 3, Wge1, g2m_edge.b1, Wge2, g2m_edge.b2, g2m_edge.g, g2m_edge.be, nullptr, msg, EG); }
  k_zero4<<<gsz(NMp * 128), 256, 0, stream>>>((float4*)agg, NMp * 128);
  k_scatter<<<gsz((long)EG * 512), 256, 0, stream>>>(msg, g2m_r, agg, (long)EG * 512);
  k_f2b<<<gsz(NMp * 512), 256, 0, stream>>>(agg, aggb, NMp * 512);
  { Chunk c[2] = {{vM, nullptr, 512}, {aggb, nullptr, 512}};
    mlp(c, 2, Wgm1, g2m_node_mesh.b1, Wgm2, g2m_node_mesh.b2,
        g2m_node_mesh.g, g2m_node_mesh.be, vM, vM, NM); }
  { Chunk c{vG, nullptr, 512};
    mlp(&c, 1, Wgg1, g2m_node_grid.b1, Wgg2, g2m_node_grid.b2,
        g2m_node_grid.g, g2m_node_grid.be, vG, vG, NG); }

  // --- processor: 4 rounds of M2M message passing ---
  for (int i = 0; i < 4; ++i) {
    Chunk ce[3] = {{vM, m2m_s, 512}, {vM, m2m_r, 512}, {eMb, nullptr, 512}};
    mlp(ce, 3, Wpe1[i], proc_edge[i].b1, Wpe2[i], proc_edge[i].b2,
        proc_edge[i].g, proc_edge[i].be, nullptr, msg, EM);
    k_addbf<<<gsz((long)EM * 256), 256, 0, stream>>>((unsigned*)eMb, (const unsigned*)msg,
                                                     (long)EM * 256);
    k_zero4<<<gsz(NMp * 128), 256, 0, stream>>>((float4*)agg, NMp * 128);
    k_scatter<<<gsz((long)EM * 512), 256, 0, stream>>>(msg, m2m_r, agg, (long)EM * 512);
    k_f2b<<<gsz(NMp * 512), 256, 0, stream>>>(agg, aggb, NMp * 512);
    Chunk cn[2] = {{vM, nullptr, 512}, {aggb, nullptr, 512}};
    mlp(cn, 2, Wpn1[i], proc_node[i].b1, Wpn2[i], proc_node[i].b2,
        proc_node[i].g, proc_node[i].be, vM, vM, NM);
  }

  // --- decoder: encode eD (reuses eXb; eG is dead), M2G msg, grid update ---
  { Chunk c{edp, nullptr, 64};
    mlp(&c, 1, Wed1, em2g_enc.b1, Wed2, em2g_enc.b2, em2g_enc.g, em2g_enc.be, nullptr, eXb, ED); }
  { Chunk c[3] = {{vM, m2g_s_p, 512}, {vG, m2g_r_p, 512}, {eXb, nullptr, 512}};
    mlp(c, 3, Wde1, m2g_edge.b1, Wde2, m2g_edge.b2, m2g_edge.g, m2g_edge.be, nullptr, msg, ED); }
  k_zero4<<<gsz(NGp * 128), 256, 0, stream>>>((float4*)agg, NGp * 128);
  k_scatter<<<gsz((long)ED * 512), 256, 0, stream>>>(msg, m2g_r, agg, (long)ED * 512);
  k_f2b<<<gsz(NGp * 512), 256, 0, stream>>>(agg, aggb, NGp * 512);
  { Chunk c[2] = {{vG, nullptr, 512}, {aggb, nullptr, 512}};
    mlp(c, 2, Wdn1, m2g_node.b1, Wdn2, m2g_node.b2, m2g_node.g, m2g_node.be, vG, vG, NG); }

  // --- out_mlp: SiLU(vG@w1+b1) @ w2 + b2 + x residual, transposed store ---
  { Chunk c{vG, nullptr, 512};
    launch_gemm(stream, 0, &c, 1, Wo1, out_b1, nullptr, nullptr, nullptr, hb, NG); }
  gemm_out64<<<dim3((unsigned)((NG + 127) / 128)), 256, 0, stream>>>(
      hb, Wo2.w, out_b2, x, (float*)d_out, NG);
}
